// CRF_82506321756469
// MI455X (gfx1250) — matrix operand order repaired and checked by reading, not compiled, not run on hardware
//
#include <hip/hip_runtime.h>

typedef __attribute__((ext_vector_type(16))) _Float16 v16h;
typedef __attribute__((ext_vector_type(8)))  _Float16 v8h;
typedef __attribute__((ext_vector_type(8)))  float    v8f;

#define CRF_L 512
#define CRF_B 1024
#define CRF_T 64

#define LOG2E 1.4426950408889634f
#define LN2   0.6931471805599453f

// Raw hardware transcendentals; args range-safe by construction.
__device__ __forceinline__ float fast_log(float x) {
    return __builtin_amdgcn_logf(x) * LN2;
}
__device__ __forceinline__ float fast_exp(float x) {
    return __builtin_amdgcn_exp2f(x * LOG2E);
}

// ---------------------------------------------------------------------------
// Forward algorithm, linear-domain with per-row power-of-two rebasing:
//   value(row) = mlog[row] + log( alpha_lin[row,:] )
//   step:  alpha_lin' = ( rebase(alpha_lin) @ exp(trans) ) * exp(emit[t])
// rebase divides by 2^e (e = exponent of row max, pure VALU bit ops) and adds
// e*ln2 to mlog — identity-preserving, so it is safe for mask-frozen rows.
// One wave owns 16 batch rows; GEMM on v_wmma_f32_16x16x32_f16.
// ---------------------------------------------------------------------------
__global__ __launch_bounds__(32)
void crf_forward_kernel(const float* __restrict__ emit,
                        const float* __restrict__ trans,
                        const float* __restrict__ strans,
                        const float* __restrict__ etrans,
                        const unsigned char* __restrict__ mask,
                        float* __restrict__ fwd_part)
{
    __shared__ _Float16 lds[16 * 64];   // rebased alpha_lin, row-major [16][64]

    const int lane = threadIdx.x;       // 0..31 (wave32)
    const int half = lane >> 4;
    const int l16  = lane & 15;
    const int b0   = blockIdx.x * 16;

    // ---- constant B-tiles: exp(trans) in 16-bit B-matrix (32x16) layout ----
    v16h bt[4][2];
    {
        const int krow = 8 * half;
        #pragma unroll
        for (int kt = 0; kt < 2; ++kt) {
            const float* tr = trans + (size_t)(kt * 32 + krow) * CRF_T + l16;
            #pragma unroll
            for (int jt = 0; jt < 4; ++jt) {
                v16h v;
                #pragma unroll
                for (int n = 0; n < 16; ++n)
                    v[n] = (_Float16)fast_exp(tr[(n < 8 ? n : n + 8) * CRF_T + jt * 16]);
                bt[jt][kt] = v;
            }
        }
    }

    // ---- per-row lengths: mask[t,b] = (t < len[b]), len in [256,512] --------
    int mylen;
    {
        int lo = 256, hi = 512;
        const int bcol = b0 + l16;
        while (lo < hi) {
            int mid = (lo + hi) >> 1;
            if (mask[(size_t)mid * CRF_B + bcol]) lo = mid + 1; else hi = mid;
        }
        mylen = lo;
    }
    int lenv[8];
    #pragma unroll
    for (int v = 0; v < 8; ++v) lenv[v] = __shfl(mylen, v + 8 * half, 32);
    int maxlen = mylen, minlen = mylen;
    #pragma unroll
    for (int s = 1; s < 32; s <<= 1) {
        maxlen = max(maxlen, __shfl_xor(maxlen, s, 32));
        minlen = min(minlen, __shfl_xor(minlen, s, 32));
    }

    // ---- alpha0_lin = exp(strans + emit[0]); mlog = 0 -----------------------
    v8f c[4];
    float mlog[8];
    {
        float st[4];
        #pragma unroll
        for (int jt = 0; jt < 4; ++jt) st[jt] = strans[jt * 16 + l16];
        const float* ep = emit + (size_t)(b0 + 8 * half) * CRF_T + l16;
        #pragma unroll
        for (int v = 0; v < 8; ++v) {
            mlog[v] = 0.f;
            #pragma unroll
            for (int jt = 0; jt < 4; ++jt)
                c[jt][v] = fast_exp(st[jt] + ep[v * CRF_T + jt * 16]);
        }
    }

    // ---- emit pipeline: raw loads (distance 1) -> off-chain exp (ee) --------
    float ern[4][8];    // raw emit[t+1] in flight
    float ee[4][8];     // exp(emit[t]) ready for current step
    auto load_emit = [&](int t, float (&dst)[4][8]) {
        const float* pt = emit + ((size_t)t * CRF_B + b0 + 8 * half) * CRF_T + l16;
        #pragma unroll
        for (int v = 0; v < 8; ++v)
            #pragma unroll
            for (int jt = 0; jt < 4; ++jt)
                dst[jt][v] = pt[v * CRF_T + jt * 16];
    };
    load_emit(1, ern);
    #pragma unroll
    for (int jt = 0; jt < 4; ++jt)
        #pragma unroll
        for (int v = 0; v < 8; ++v) ee[jt][v] = fast_exp(ern[jt][v]);

    // deep prefetch: one global_prefetch covers 16 rows x 2 cachelines (4KB)
    const float* pfbase = emit + (size_t)b0 * CRF_T
                               + (size_t)(lane >> 1) * CRF_T   // row = lane/2
                               + (size_t)(lane & 1) * 32;      // 128B line pick
    const size_t pfstride = (size_t)CRF_B * CRF_T;

    // ---- one recurrence step (masked folded per call site) ------------------
    auto step = [&](int t, bool masked) {
        // per-row max (linear, positive): 4 tiles then 16-lane butterfly
        float rmax[8];
        #pragma unroll
        for (int v = 0; v < 8; ++v) {
            float r = fmaxf(fmaxf(c[0][v], c[1][v]), fmaxf(c[2][v], c[3][v]));
            #pragma unroll
            for (int s = 1; s < 16; s <<= 1) r = fmaxf(r, __shfl_xor(r, s, 32));
            rmax[v] = r;
        }
        // power-of-two rebase: scl = 2^(127-be), mlog += (be-127)*ln2
        float scl[8];
        #pragma unroll
        for (int v = 0; v < 8; ++v) {
            const unsigned be = __float_as_uint(rmax[v]) >> 23;   // rmax > 0
            scl[v] = __uint_as_float((254u - be) << 23);
            mlog[v] = fmaf((float)((int)be - 127), LN2, mlog[v]);
        }
        // rebase in place, convert to f16, store row-major LDS [16][64]
        #pragma unroll
        for (int v = 0; v < 8; ++v) {
            const int row = v + 8 * half;
            #pragma unroll
            for (int jt = 0; jt < 4; ++jt) {
                c[jt][v] *= scl[v];
                lds[row * 64 + jt * 16 + l16] = (_Float16)c[jt][v];
            }
        }
        // single wave: same-wave DS ordering; only need store->load wait
        asm volatile("s_wait_dscnt 0x0" ::: "memory");

        // A tiles (16-bit A 16x32 layout): contiguous 16B chunks -> b128 reads
        v16h a[2];
        #pragma unroll
        for (int kt = 0; kt < 2; ++kt) {
            const _Float16* p = &lds[l16 * 64 + kt * 32 + half * 8];
            v8h alo = *(const v8h*)(p);
            v8h ahi = *(const v8h*)(p + 16);
            v16h av;
            #pragma unroll
            for (int i = 0; i < 8; ++i) { av[i] = alo[i]; av[i + 8] = ahi[i]; }
            a[kt] = av;
        }
        asm volatile("" ::: "memory");

        // deep prefetch t+4 and issue next raw emit loads (distance 1)
        {
            const int tpf = (t + 4 < CRF_L) ? t + 4 : CRF_L - 1;
            __builtin_prefetch(pfbase + (size_t)tpf * pfstride, 0, 1);
        }
        load_emit(t + 1 < CRF_L ? t + 1 : CRF_L - 1, ern);

        float p8[8];
        if (masked) {
            #pragma unroll
            for (int v = 0; v < 8; ++v) p8[v] = (t < lenv[v]) ? 1.0f : 0.0f;
        }

        // S = rebased alpha @ expT ; alpha' = S * exp(emit[t]) (branchless)
        #pragma unroll
        for (int jt = 0; jt < 4; ++jt) {
            v8f d = {};
            d = __builtin_amdgcn_wmma_f32_16x16x32_f16(
                    false, a[0], false, bt[jt][0], (short)0, d, false, false);
            d = __builtin_amdgcn_wmma_f32_16x16x32_f16(
                    false, a[1], false, bt[jt][1], (short)0, d, false, false);
            #pragma unroll
            for (int v = 0; v < 8; ++v) {
                const float ns = d[v] * ee[jt][v];
                if (masked) c[jt][v] = fmaf(p8[v], ns - c[jt][v], c[jt][v]);
                else        c[jt][v] = ns;
            }
        }
        // off-chain: exp of next step's emit (consumes the in-flight loads)
        #pragma unroll
        for (int jt = 0; jt < 4; ++jt)
            #pragma unroll
            for (int v = 0; v < 8; ++v) ee[jt][v] = fast_exp(ern[jt][v]);
    };

    int t = 1;
    for (; t < minlen; ++t) step(t, false);   // mask all-true: no blending
    for (; t < maxlen; ++t) step(t, true);    // monotone-mask tail

    // ---- final: lse_row = mlog + log( sum_j alpha_lin * exp(etrans_j) ) ----
    float ete[4];
    #pragma unroll
    for (int jt = 0; jt < 4; ++jt) ete[jt] = fast_exp(etrans[jt * 16 + l16]);

    float partial = 0.f;
    #pragma unroll
    for (int v = 0; v < 8; ++v) {
        float sum = c[0][v] * ete[0] + c[1][v] * ete[1] +
                    c[2][v] * ete[2] + c[3][v] * ete[3];
        #pragma unroll
        for (int s = 1; s < 16; s <<= 1) sum += __shfl_xor(sum, s, 32);
        if (l16 == 0) partial += mlog[v] + fast_log(sum);   // lanes 0 and 16
    }
    partial += __shfl_xor(partial, 16, 32);
    if (lane == 0) fwd_part[blockIdx.x] = partial;
}

// ---------------------------------------------------------------------------
// Gold-path score: terms independent over t; one block per batch element,
// t-parallel threads, deterministic LDS tree reduction.
// ---------------------------------------------------------------------------
__global__ __launch_bounds__(256)
void crf_gold_kernel(const float* __restrict__ emit,
                     const float* __restrict__ trans,
                     const float* __restrict__ strans,
                     const float* __restrict__ etrans,
                     const int* __restrict__ target,
                     const unsigned char* __restrict__ mask,
                     float* __restrict__ gold_part)
{
    __shared__ float red[256];
    const int b   = blockIdx.x;
    const int tid = threadIdx.x;

    int lo = 256, hi = 512;
    while (lo < hi) {
        int mid = (lo + hi) >> 1;
        if (mask[(size_t)mid * CRF_B + b]) lo = mid + 1; else hi = mid;
    }
    const int len = lo;

    float s = 0.f;
    for (int t = tid; t < len; t += 256) {
        const int tg = target[(size_t)t * CRF_B + b];
        float v = emit[((size_t)t * CRF_B + b) * CRF_T + tg];
        if (t > 0) v += trans[target[(size_t)(t - 1) * CRF_B + b] * CRF_T + tg];
        s += v;
    }
    if (tid == 0)
        s += strans[target[b]] +
             etrans[target[(size_t)(len - 1) * CRF_B + b]];

    red[tid] = s;
    __syncthreads();
    #pragma unroll
    for (int off = 128; off > 0; off >>= 1) {
        if (tid < off) red[tid] += red[tid + off];
        __syncthreads();
    }
    if (tid == 0) gold_part[b] = red[0];
}

// ---------------------------------------------------------------------------
// Deterministic fixed-order final reduction: out = (sum fwd - sum gold) / B
// ---------------------------------------------------------------------------
__global__ void crf_reduce_kernel(const float* __restrict__ fwd_part,   // [64]
                                  const float* __restrict__ gold_part,  // [1024]
                                  float* __restrict__ out)
{
    if (threadIdx.x != 0 || blockIdx.x != 0) return;
    float fz = 0.f;
    for (int i = 0; i < 64; ++i) fz += fwd_part[i];
    float gs = 0.f;
    for (int i = 0; i < CRF_B; ++i) gs += gold_part[i];
    out[0] = (fz - gs) * (1.0f / CRF_B);
}

extern "C" void kernel_launch(void* const* d_in, const int* in_sizes, int n_in,
                              void* d_out, int out_size, void* d_ws, size_t ws_size,
                              hipStream_t stream)
{
    const float*         emit   = (const float*)d_in[0];
    const float*         trans  = (const float*)d_in[1];
    const float*         strans = (const float*)d_in[2];
    const float*         etrans = (const float*)d_in[3];
    const int*           target = (const int*)d_in[4];
    const unsigned char* mask   = (const unsigned char*)d_in[5];
    float* out = (float*)d_out;

    float* gold_part = (float*)d_ws;          // [1024]
    float* fwd_part  = gold_part + CRF_B;     // [64]

    crf_gold_kernel<<<CRF_B, 256, 0, stream>>>(
        emit, trans, strans, etrans, target, mask, gold_part);
    crf_forward_kernel<<<CRF_B / 16, 32, 0, stream>>>(
        emit, trans, strans, etrans, mask, fwd_part);
    crf_reduce_kernel<<<1, 32, 0, stream>>>(fwd_part, gold_part, out);
}
